// CBOWModel_85194971283909
// MI455X (gfx1250) — matrix operand order repaired and checked by reading, not compiled, not run on hardware
//
#include <hip/hip_runtime.h>
#include <hip/hip_bf16.h>

// Problem constants (match reference setup_inputs)
#define VOCAB 100000
#define DIM   128
#define BATCHN 16384
#define CTXW  8
#define KNEG  10

// Tiling
#define TILE  16                 // batch rows per wave (WMMA M/N)
#define NWAVE 4                  // waves per block
#define BLOCKT (NWAVE * 32)      // wave32
#define LROW  (DIM + 4)          // LDS row padding: 4-bank stride per row -> conflict-free

typedef __attribute__((ext_vector_type(16))) _Float16 v16h;
typedef __attribute__((ext_vector_type(8)))  float    v8f;

// stable softplus: max(x,0) + log(1 + exp(-|x|)), hw transcendentals only
__device__ __forceinline__ float softplus_f(float x) {
    return fmaxf(x, 0.0f) + __logf(1.0f + __expf(-fabsf(x)));
}

// pack two float4 -> 8 consecutive f16 elements of a v16h
#define PACK8(dst, off, A, B) do {                                          \
    (dst)[(off)+0] = (_Float16)(A).x; (dst)[(off)+1] = (_Float16)(A).y;     \
    (dst)[(off)+2] = (_Float16)(A).z; (dst)[(off)+3] = (_Float16)(A).w;     \
    (dst)[(off)+4] = (_Float16)(B).x; (dst)[(off)+5] = (_Float16)(B).y;     \
    (dst)[(off)+6] = (_Float16)(B).z; (dst)[(off)+7] = (_Float16)(B).w; } while (0)

__global__ __launch_bounds__(BLOCKT)
void cbow_nsloss_wmma_kernel(const int*   __restrict__ center,    // [B]
                             const int*   __restrict__ context,   // [B][CTXW]
                             const int*   __restrict__ negctx,    // [B][KNEG]
                             const float* __restrict__ inW,       // [VOCAB][DIM]
                             const float* __restrict__ outW,      // [VOCAB][DIM]
                             float*       __restrict__ out)       // [1]
{
    __shared__ __align__(16) float s_ctx[NWAVE][TILE][LROW];

    const int lane     = threadIdx.x & 31;
    const int wave     = threadIdx.x >> 5;
    const int tileBase = (blockIdx.x * NWAVE + wave) * TILE;
    float (*ctx)[LROW] = s_ctx[wave];

    // ---------- Phase 1: gather + mean context embeddings into LDS (f32) ----------
    // 32 lanes x float4 = one full 128-float row per iteration; fully coalesced.
    const float invC = 1.0f / (float)CTXW;
    for (int r = 0; r < TILE; ++r) {
        const int b = tileBase + r;              // wave-uniform -> indices via SMEM
        float ax = 0.f, ay = 0.f, az = 0.f, aw = 0.f;
        #pragma unroll
        for (int c = 0; c < CTXW; ++c) {
            const int idx = context[b * CTXW + c];
            const float4 v = ((const float4*)(inW + (size_t)idx * DIM))[lane];
            ax += v.x; ay += v.y; az += v.z; aw += v.w;
        }
        *(float4*)(&ctx[r][lane * 4]) = make_float4(ax * invC, ay * invC, az * invC, aw * invC);
    }
    // Same-wave LDS RAW: DS ops are in-order per wave; compiler inserts s_wait_dscnt.

    // ---------- Phase 2: build A operand (16x32 f16 per K-slab) from LDS ----------
    // ISA A layout: lanes 0-15 (M=lane): elems = K {0..7, 16..23}; lanes 16-31: {8..15, 24..31}
    const int mLane = lane & 15;
    const int hi    = lane >> 4;

    v16h aS[4];
    #pragma unroll
    for (int s = 0; s < 4; ++s) {
        const int k0 = s * 32 + hi * 8;
        const float* cr = &ctx[mLane][0];
        const float4 p0 = *(const float4*)(cr + k0);
        const float4 p1 = *(const float4*)(cr + k0 + 4);
        const float4 p2 = *(const float4*)(cr + k0 + 16);
        const float4 p3 = *(const float4*)(cr + k0 + 20);
        v16h a;
        PACK8(a, 0, p0, p1);
        PACK8(a, 8, p2, p3);
        aS[s] = a;
    }

    // ---------- Phase 3: 11 targets (1 center + 10 negatives) ----------
    const int  bLane = tileBase + mLane;
    const int  cIdx  = center[bLane];
    const bool validLane = (lane < 8) || (lane >= 24);   // lanes holding a diagonal elem
    const int  dsel  = lane & 7;                          // acc index of that elem

    float localLoss = 0.0f;

    for (int j = 0; j < 1 + KNEG; ++j) {
        const int gidx = (j == 0) ? cIdx : negctx[bLane * KNEG + (j - 1)];
        const float* trow = outW + (size_t)gidx * DIM;

        v8f acc = {};
        #pragma unroll
        for (int s = 0; s < 4; ++s) {
            // ISA B layout (32x16): lane n=lane&15, elems = K {hi*8 .. hi*8+7, 16+hi*8 .. 16+hi*8+7} of slab s
            const int k0 = s * 32 + hi * 8;
            const float4 q0 = *(const float4*)(trow + k0);
            const float4 q1 = *(const float4*)(trow + k0 + 4);
            const float4 q2 = *(const float4*)(trow + k0 + 16);
            const float4 q3 = *(const float4*)(trow + k0 + 20);
            v16h bm;
            PACK8(bm, 0, q0, q1);
            PACK8(bm, 8, q2, q3);
            // D = A*B + C, f32 accumulate; EXEC is all-ones here (no divergence)
            acc = __builtin_amdgcn_wmma_f32_16x16x32_f16(
                false, aS[s], false, bm, (short)0, acc, false, false);
        }

        // Diagonal extraction: m<8 -> lane m, acc[m]; m>=8 -> lane m+16, acc[m-8]
        float x = acc[0];
        #pragma unroll
        for (int q = 1; q < 8; ++q) x = (dsel == q) ? acc[q] : x;

        const float contrib = (j == 0) ? softplus_f(-x) : softplus_f(x);
        localLoss += validLane ? contrib : 0.0f;   // select, keeps EXEC uniform
    }

    // ---------- Phase 4: wave reduce + single atomic per wave ----------
    #pragma unroll
    for (int off = 16; off > 0; off >>= 1)
        localLoss += __shfl_xor(localLoss, off, 32);
    if (lane == 0)
        atomicAdd(out, localLoss * (1.0f / (float)BATCHN));
}

extern "C" void kernel_launch(void* const* d_in, const int* in_sizes, int n_in,
                              void* d_out, int out_size, void* d_ws, size_t ws_size,
                              hipStream_t stream) {
    (void)in_sizes; (void)n_in; (void)out_size; (void)d_ws; (void)ws_size;
    const int*   center  = (const int*)d_in[0];
    const int*   context = (const int*)d_in[1];
    const int*   negctx  = (const int*)d_in[2];
    const float* inW     = (const float*)d_in[3];
    const float* outW    = (const float*)d_in[4];
    float*       out     = (float*)d_out;

    // d_out is poisoned; zero the scalar accumulator (graph-capture safe)
    hipMemsetAsync(out, 0, sizeof(float), stream);

    const int tiles  = BATCHN / TILE;          // 1024
    const int blocks = tiles / NWAVE;          // 256
    cbow_nsloss_wmma_kernel<<<blocks, BLOCKT, 0, stream>>>(
        center, context, negctx, inW, outW, out);
}
